// attention_20091857011517
// MI455X (gfx1250) — matrix operand order repaired and checked by reading, not compiled, not run on hardware
//
#include <hip/hip_runtime.h>
#include <hip/hip_bf16.h>
#include <math.h>

typedef __bf16 bf16_t;
typedef __attribute__((ext_vector_type(16))) __bf16 v16bf;
typedef __attribute__((ext_vector_type(8)))  __bf16 v8bf;
typedef __attribute__((ext_vector_type(8)))  float  v8f;

#define DIMS 2048
#define HEAD 16
#define HD   128
#define HD2  64
#define SEQ  2048

static __device__ __forceinline__ v16bf cat8(v8bf lo, v8bf hi) {
  return __builtin_shufflevector(lo, hi, 0,1,2,3,4,5,6,7,8,9,10,11,12,13,14,15);
}

static __device__ __forceinline__ v8f wmma_bf16(v16bf a, v16bf b, v8f c) {
  // D = A(16x32 bf16) * B(32x16 bf16) + C(16x16 f32)
  return __builtin_amdgcn_wmma_f32_16x16x32_bf16(false, a, false, b, (short)0, c,
                                                 false, false);
}

static __device__ __forceinline__ v8f zero8() {
  v8f z = {0.f,0.f,0.f,0.f,0.f,0.f,0.f,0.f};
  return z;
}

// CDNA5 async global->LDS copy (ASYNCcnt-tracked).  lds_off = wave-relative LDS
// byte offset, gaddr = 64-bit global address.  16 bytes per issue.
static __device__ __forceinline__ void async_copy_b128(unsigned lds_off,
                                                       const void* gptr) {
  unsigned long long ga = (unsigned long long)(uintptr_t)gptr;
  asm volatile("global_load_async_to_lds_b128 %0, %1, off"
               :: "v"(lds_off), "v"(ga) : "memory");
}

static __device__ __forceinline__ void wait_async_le2() {
  asm volatile("s_wait_asynccnt 0x2" ::: "memory");
}
static __device__ __forceinline__ void wait_async_0() {
  asm volatile("s_wait_asynccnt 0x0" ::: "memory");
}

// ---------------------------------------------------------------------------
// Transpose fp32 W[K][N] -> bf16 WT[N][K]
// ---------------------------------------------------------------------------
__global__ void k_transpose_bf16(const float* __restrict__ W, bf16_t* __restrict__ WT,
                                 int K, int N) {
  size_t idx = (size_t)blockIdx.x * blockDim.x + threadIdx.x;
  size_t total = (size_t)K * N;
  if (idx >= total) return;
  size_t k = idx / N, n = idx - k * (size_t)N;
  WT[n * (size_t)K + k] = (bf16_t)W[idx];
}

// ---------------------------------------------------------------------------
// Per-token LayerNorm stats (shared between LNq / LNkv) + magnitude ||x||.
// ---------------------------------------------------------------------------
__global__ void k_ln_mag(const float* __restrict__ x,
                         const float* __restrict__ gq, const float* __restrict__ bq,
                         const float* __restrict__ gkv, const float* __restrict__ bkv,
                         bf16_t* __restrict__ xq, bf16_t* __restrict__ xkv,
                         float* __restrict__ mag) {
  int c = blockIdx.x, t = threadIdx.x;
  const float* row = x + (size_t)c * DIMS;
  float s = 0.f, ss = 0.f;
  for (int i = t; i < DIMS; i += 256) { float v = row[i]; s += v; ss += v * v; }
  __shared__ float rs[256], rss[256];
  rs[t] = s; rss[t] = ss;
  __syncthreads();
  for (int o = 128; o > 0; o >>= 1) {
    if (t < o) { rs[t] += rs[t + o]; rss[t] += rss[t + o]; }
    __syncthreads();
  }
  float mean = rs[0] * (1.0f / DIMS);
  float var  = rss[0] * (1.0f / DIMS) - mean * mean;
  float rstd = rsqrtf(var + 1e-5f);
  if (t == 0) mag[c] = sqrtf(rss[0]);
  for (int i = t; i < DIMS; i += 256) {
    float nv = (row[i] - mean) * rstd;
    xq [(size_t)c * DIMS + i] = (bf16_t)(nv * gq[i]  + bq[i]);
    xkv[(size_t)c * DIMS + i] = (bf16_t)(nv * gkv[i] + bkv[i]);
  }
}

// ---------------------------------------------------------------------------
// Register-blocked bf16 WMMA GEMM with async-to-LDS A staging.
//   C[M][N] (f32) = A[M][K] (bf16) * BT[N][K]^T + bias[N]
// Block tile 128x256, 8 waves = 2(m) x 4(n), wave tile 64x64 (4x4 WMMA tiles).
// A chunk (128x32 bf16 = 8KB) double-buffered in LDS via
// global_load_async_to_lds_b128; B fragments direct from global (contiguous).
// ---------------------------------------------------------------------------
__global__ void __launch_bounds__(256)
k_gemm_bf16(const bf16_t* __restrict__ A, const bf16_t* __restrict__ BT,
            const float* __restrict__ bias, float* __restrict__ Cout,
            int M, int N, int K) {
  __shared__ __bf16 asmem[2][128 * 32];
  int tid = threadIdx.x;
  int wid = tid >> 5, lane = tid & 31;
  int wm = wid >> 2, wn = wid & 3;            // wave grid 2 x 4 inside block
  int half = lane >> 4, lm = lane & 15;

  int nBlocksN = N >> 8;
  int bm = blockIdx.x / nBlocksN, bn = blockIdx.x - bm * nBlocksN;
  int rowBase = bm * 128;
  int colBase = bn * 256 + wn * 64;
  if (rowBase >= M) return;

  // Async A staging: 512 x 16B segments per K-step; 2 per thread.
  int r0 = tid >> 2,          c0 = (tid & 3) * 8;
  int r1 = (tid + 256) >> 2,  c1 = ((tid + 256) & 3) * 8;
  const bf16_t* a0p = A + (size_t)(rowBase + r0) * K + c0;
  const bf16_t* a1p = A + (size_t)(rowBase + r1) * K + c1;
  unsigned l0 = (unsigned)(uintptr_t)(&asmem[0][r0 * 32 + c0]);
  unsigned l1 = (unsigned)(uintptr_t)(&asmem[0][r1 * 32 + c1]);

  // B fragment row pointers: col = colBase + nt*16 + lm, K-base = half*16
  const bf16_t* brow[4];
  #pragma unroll
  for (int nt = 0; nt < 4; ++nt)
    brow[nt] = BT + (size_t)(colBase + nt * 16 + lm) * K + half * 16;

  v8f acc[4][4];
  #pragma unroll
  for (int mt = 0; mt < 4; ++mt)
    #pragma unroll
    for (int nt = 0; nt < 4; ++nt) acc[mt][nt] = zero8();

  int nk = K >> 5;
  // prime buffer 0
  async_copy_b128(l0, a0p);
  async_copy_b128(l1, a1p);

  for (int kt = 0; kt < nk; ++kt) {
    int cur = kt & 1;
    int k = kt << 5;
    if (kt + 1 < nk) {                        // prefetch next K-chunk into other buf
      unsigned off = (cur ^ 1) ? 8192u : 0u;
      async_copy_b128(l0 + off, a0p + k + 32);
      async_copy_b128(l1 + off, a1p + k + 32);
      wait_async_le2();                       // oldest 2 (current buf) complete
    } else {
      wait_async_0();
    }
    __syncthreads();                          // all waves' A chunk visible

    const __bf16* abase = &asmem[cur][0];
    v16bf afr[4], bfr[4];
    #pragma unroll
    for (int mt = 0; mt < 4; ++mt) {
      const __bf16* ap = abase + (wm * 64 + mt * 16 + lm) * 32 + half * 16;
      afr[mt] = cat8(*(const v8bf*)ap, *(const v8bf*)(ap + 8));
    }
    #pragma unroll
    for (int nt = 0; nt < 4; ++nt) {
      const bf16_t* bp = brow[nt] + k;
      __builtin_prefetch(bp + 256, 0, 0);
      bfr[nt] = cat8(*(const v8bf*)bp, *(const v8bf*)(bp + 8));
    }
    #pragma unroll
    for (int mt = 0; mt < 4; ++mt)
      #pragma unroll
      for (int nt = 0; nt < 4; ++nt)
        acc[mt][nt] = wmma_bf16(afr[mt], bfr[nt], acc[mt][nt]);

    __syncthreads();                          // done reading buf before overwrite
  }

  // C/D layout: VGPR r -> row = r + 8*half, col = lm
  #pragma unroll
  for (int nt = 0; nt < 4; ++nt) {
    int col = colBase + nt * 16 + lm;
    float bcol = bias[col];
    #pragma unroll
    for (int mt = 0; mt < 4; ++mt) {
      int row = rowBase + wm * 64 + mt * 16 + 8 * half;
      #pragma unroll
      for (int r = 0; r < 8; ++r)
        Cout[(size_t)(row + r) * N + col] = acc[mt][nt][r] + bcol;
    }
  }
}

// ---------------------------------------------------------------------------
// scale * rotary(mag-polar) * head-LN. Emits qn,kn as [H][C][HD] bf16 and
// V transposed as [H][HD][C] bf16. One block per (token, head), 64 pair-threads.
// ---------------------------------------------------------------------------
__global__ void k_rope_headln(const float* __restrict__ q,   // [SEQ][DIMS]
                              const float* __restrict__ kv,  // [SEQ][2*DIMS]
                              const float* __restrict__ mag, // [SEQ]
                              const float* __restrict__ gh, const float* __restrict__ bh,
                              bf16_t* __restrict__ qn, bf16_t* __restrict__ kn,
                              bf16_t* __restrict__ vT) {
  int c = blockIdx.x, h = blockIdx.y, p = threadIdx.x;  // p = rotary pair [0,64)
  const float scale = 0.29730177875068026f;             // 128^-0.25
  float fr  = 0.04f * __powf(40.0f, (float)p * (1.0f / 63.0f));
  float ang = (float)c * fr;
  float m = mag[c];
  float rr = m * __cosf(ang), ri = m * __sinf(ang);

  size_t qoff = (size_t)c * DIMS + (size_t)h * HD + 2 * p;
  float qr = q[qoff] * scale, qi = q[qoff + 1] * scale;
  float q_r = qr * rr - qi * ri, q_i = qr * ri + qi * rr;

  size_t koff = (size_t)c * (2 * DIMS) + (size_t)h * HD + 2 * p;
  float kr = kv[koff] * scale, ki = kv[koff + 1] * scale;
  float k_r = kr * rr - ki * ri, k_i = kr * ri + ki * rr;

  __shared__ float sb[64][4];
  __shared__ float tot[4];
  sb[p][0] = q_r + q_i; sb[p][1] = q_r * q_r + q_i * q_i;
  sb[p][2] = k_r + k_i; sb[p][3] = k_r * k_r + k_i * k_i;
  __syncthreads();
  if (p == 0) {
    float t0 = 0, t1 = 0, t2 = 0, t3 = 0;
    for (int i = 0; i < 64; ++i) { t0 += sb[i][0]; t1 += sb[i][1]; t2 += sb[i][2]; t3 += sb[i][3]; }
    tot[0] = t0; tot[1] = t1; tot[2] = t2; tot[3] = t3;
  }
  __syncthreads();
  float qmu = tot[0] * (1.0f / HD), qvar = tot[1] * (1.0f / HD) - qmu * qmu;
  float kmu = tot[2] * (1.0f / HD), kvar = tot[3] * (1.0f / HD) - kmu * kmu;
  float qrstd = rsqrtf(qvar + 1e-5f), krstd = rsqrtf(kvar + 1e-5f);
  float g = gh[h], bb = bh[h];  // per-head affine

  size_t base = ((size_t)h * SEQ + c) * HD + 2 * p;
  qn[base]     = (bf16_t)((q_r - qmu) * qrstd * g + bb);
  qn[base + 1] = (bf16_t)((q_i - qmu) * qrstd * g + bb);
  kn[base]     = (bf16_t)((k_r - kmu) * krstd * g + bb);
  kn[base + 1] = (bf16_t)((k_i - kmu) * krstd * g + bb);

  size_t voff = (size_t)c * (2 * DIMS) + DIMS + (size_t)h * HD + 2 * p;
  vT[((size_t)h * HD + 2 * p    ) * SEQ + c] = (bf16_t)kv[voff];
  vT[((size_t)h * HD + 2 * p + 1) * SEQ + c] = (bf16_t)kv[voff + 1];
}

// ---------------------------------------------------------------------------
// Flash attention: one wave = one (head, 16-row q tile). Online softmax over
// 32-token k chunks. S and P*V both via v_wmma_f32_16x16x32_bf16.
// ---------------------------------------------------------------------------
__global__ void __launch_bounds__(256)
k_attn(const bf16_t* __restrict__ qn, const bf16_t* __restrict__ kn,
       const bf16_t* __restrict__ vT, bf16_t* __restrict__ aout) {
  __shared__ __bf16 pbuf[8 * 16 * 32];       // per-wave P staging tiles
  int wid  = threadIdx.x >> 5;
  int lane = threadIdx.x & 31;
  int wave = blockIdx.x * 8 + wid;
  int h  = wave >> 7;                        // / (SEQ/16)
  int qt = wave & 127;
  int half = lane >> 4, lm = lane & 15;
  const float lsc = 0.08838834764831845f;    // 1/sqrt(HD)

  v16bf aq[4];
  {
    const bf16_t* qrow = qn + ((size_t)h * SEQ + qt * 16 + lm) * HD + half * 16;
    #pragma unroll
    for (int kc = 0; kc < 4; ++kc)
      aq[kc] = cat8(*(const v8bf*)(qrow + kc * 32),
                    *(const v8bf*)(qrow + kc * 32 + 8));
  }

  v8f o[8];
  #pragma unroll
  for (int i = 0; i < 8; ++i) o[i] = zero8();
  float runM[8], runS[8];
  #pragma unroll
  for (int r = 0; r < 8; ++r) { runM[r] = -1e30f; runS[r] = 0.f; }

  __bf16* pl = pbuf + wid * 512;

  for (int kt = 0; kt < SEQ / 32; ++kt) {
    v8f s0 = zero8(), s1 = zero8();
    const bf16_t* kbase = kn + ((size_t)h * SEQ + kt * 32) * HD;
    #pragma unroll
    for (int kc = 0; kc < 4; ++kc) {
      const bf16_t* kp0 = kbase + (size_t)lm * HD + kc * 32 + half * 16;
      s0 = wmma_bf16(aq[kc], cat8(*(const v8bf*)kp0, *(const v8bf*)(kp0 + 8)), s0);
      const bf16_t* kp1 = kbase + (size_t)(16 + lm) * HD + kc * 32 + half * 16;
      s1 = wmma_bf16(aq[kc], cat8(*(const v8bf*)kp1, *(const v8bf*)(kp1 + 8)), s1);
    }
    #pragma unroll
    for (int r = 0; r < 8; ++r) {
      float x0 = s0[r] * lsc, x1 = s1[r] * lsc;
      float mx = fmaxf(x0, x1);
      #pragma unroll
      for (int msk = 8; msk >= 1; msk >>= 1)
        mx = fmaxf(mx, __shfl_xor(mx, msk, 32));
      float newM  = fmaxf(runM[r], mx);
      float alpha = __expf(runM[r] - newM);
      float p0 = __expf(x0 - newM), p1 = __expf(x1 - newM);
      float ps = p0 + p1;
      #pragma unroll
      for (int msk = 8; msk >= 1; msk >>= 1)
        ps += __shfl_xor(ps, msk, 32);
      runS[r] = runS[r] * alpha + ps;
      runM[r] = newM;
      #pragma unroll
      for (int nt = 0; nt < 8; ++nt) o[nt][r] *= alpha;
      int m = r + 8 * half;
      pl[m * 32 + lm]      = (__bf16)p0;
      pl[m * 32 + 16 + lm] = (__bf16)p1;
    }
    const __bf16* pr = pl + lm * 32 + half * 16;
    v16bf pa = cat8(*(const v8bf*)pr, *(const v8bf*)(pr + 8));
    const bf16_t* vbase = vT + (size_t)h * HD * SEQ + kt * 32 + half * 16;
    #pragma unroll
    for (int nt = 0; nt < 8; ++nt) {
      const bf16_t* vp = vbase + (size_t)(nt * 16 + lm) * SEQ;
      o[nt] = wmma_bf16(pa, cat8(*(const v8bf*)vp, *(const v8bf*)(vp + 8)), o[nt]);
    }
  }
  #pragma unroll
  for (int r = 0; r < 8; ++r) {
    int m = r + 8 * half;
    float inv = 1.0f / runS[r];
    size_t rowoff = (size_t)(qt * 16 + m) * DIMS + (size_t)h * HD + lm;
    #pragma unroll
    for (int nt = 0; nt < 8; ++nt)
      aout[rowoff + nt * 16] = (bf16_t)(o[nt][r] * inv);
  }
}

// ---------------------------------------------------------------------------
extern "C" void kernel_launch(void* const* d_in, const int* in_sizes, int n_in,
                              void* d_out, int out_size, void* d_ws, size_t ws_size,
                              hipStream_t stream) {
  const float* x      = (const float*)d_in[0];
  const float* ln_q_g = (const float*)d_in[1];
  const float* ln_q_b = (const float*)d_in[2];
  const float* Wq     = (const float*)d_in[3];
  const float* bq     = (const float*)d_in[4];
  const float* ln_kv_g= (const float*)d_in[5];
  const float* ln_kv_b= (const float*)d_in[6];
  const float* Wkv    = (const float*)d_in[7];
  const float* bkv    = (const float*)d_in[8];
  const float* ln_h_g = (const float*)d_in[9];
  const float* ln_h_b = (const float*)d_in[10];
  const float* Wo     = (const float*)d_in[11];
  const float* bo     = (const float*)d_in[12];
  float* out = (float*)d_out;

  char* w = (char*)d_ws;
  auto carve = [&](size_t bytes) -> char* {
    char* p = w; w += (bytes + 255) & ~(size_t)255; return p;
  };
  bf16_t* WqT  = (bf16_t*)carve((size_t)DIMS * DIMS * 2);
  bf16_t* WkvT = (bf16_t*)carve((size_t)DIMS * 2 * DIMS * 2);
  bf16_t* WoT  = (bf16_t*)carve((size_t)DIMS * DIMS * 2);
  bf16_t* xq   = (bf16_t*)carve((size_t)SEQ * DIMS * 2);
  bf16_t* xkv  = (bf16_t*)carve((size_t)SEQ * DIMS * 2);
  float*  mag  = (float*) carve((size_t)SEQ * 4);
  float*  qbuf = (float*) carve((size_t)SEQ * DIMS * 4);
  float*  kvbuf= (float*) carve((size_t)SEQ * 2 * DIMS * 4);
  bf16_t* qnb  = (bf16_t*)carve((size_t)HEAD * SEQ * HD * 2);
  bf16_t* knb  = (bf16_t*)carve((size_t)HEAD * SEQ * HD * 2);
  bf16_t* vTb  = (bf16_t*)carve((size_t)HEAD * HD * SEQ * 2);
  bf16_t* abuf = (bf16_t*)carve((size_t)SEQ * DIMS * 2);

  const int thr = 256;
  k_transpose_bf16<<<(DIMS * DIMS + thr - 1) / thr, thr, 0, stream>>>(Wq, WqT, DIMS, DIMS);
  k_transpose_bf16<<<(DIMS * 2 * DIMS + thr - 1) / thr, thr, 0, stream>>>(Wkv, WkvT, DIMS, 2 * DIMS);
  k_transpose_bf16<<<(DIMS * DIMS + thr - 1) / thr, thr, 0, stream>>>(Wo, WoT, DIMS, DIMS);
  k_ln_mag<<<SEQ, 256, 0, stream>>>(x, ln_q_g, ln_q_b, ln_kv_g, ln_kv_b, xq, xkv, mag);
  // block tile 128x256
  k_gemm_bf16<<<(SEQ / 128) * (DIMS / 256), thr, 0, stream>>>(xq, WqT, bq, qbuf, SEQ, DIMS, DIMS);
  k_gemm_bf16<<<(SEQ / 128) * (2 * DIMS / 256), thr, 0, stream>>>(xkv, WkvT, bkv, kvbuf, SEQ, 2 * DIMS, DIMS);
  k_rope_headln<<<dim3(SEQ, HEAD), 64, 0, stream>>>(qbuf, kvbuf, mag, ln_h_g, ln_h_b, qnb, knb, vTb);
  k_attn<<<HEAD * (SEQ / 16) / 8, thr, 0, stream>>>(qnb, knb, vTb, abuf);
  k_gemm_bf16<<<(SEQ / 128) * (DIMS / 256), thr, 0, stream>>>(abuf, WoT, bo, out, SEQ, DIMS, DIMS);
}
